// GAT_1065151890036
// MI455X (gfx1250) — matrix operand order repaired and checked by reading, not compiled, not run on hardware
//
#include <hip/hip_runtime.h>
#include <hip/hip_bf16.h>

typedef _Float16 v16h __attribute__((ext_vector_type(16)));
typedef float    v8f  __attribute__((ext_vector_type(8)));
typedef unsigned int u32x4 __attribute__((ext_vector_type(4)));
typedef int      v4i_ __attribute__((ext_vector_type(4)));

#define NIN   768
#define HOUT  768          // HEADS*HID
#define HEADS 8
#define HID   96
#define NCLS  3
#define NEG_SLOPE 0.2f
#define ENC_NEG_INF 0x007FFFFFu
#define KTILES (NIN / 32)

#if defined(__has_builtin)
#if __has_builtin(__builtin_amdgcn_global_load_async_to_lds_b128)
#define GAT_ASYNC_LDS 1
#endif
#endif

// ---------- helpers ----------
__device__ inline unsigned fenc(float f) {
  unsigned u = __float_as_uint(f);
  return (u & 0x80000000u) ? ~u : (u | 0x80000000u);
}
__device__ inline float fdec(unsigned u) {
  unsigned v = (u & 0x80000000u) ? (u & 0x7FFFFFFFu) : ~u;
  return __uint_as_float(v);
}
__device__ inline void edge_sd(const int* ei, long long e, long long E, int& s, int& d) {
  if (e < E) { s = ei[e]; d = ei[E + e]; }
  else       { s = d = (int)(e - E); }
}

// 16-byte global->LDS copy: async on CDNA5, sync fallback otherwise.
__device__ __forceinline__ void cp16_g2s(void* lds, const void* g) {
#ifdef GAT_ASYNC_LDS
  __builtin_amdgcn_global_load_async_to_lds_b128(
      (__attribute__((address_space(1))) v4i_*)(g),
      (__attribute__((address_space(3))) v4i_*)(lds), 0, 0);
#else
  *(u32x4*)lds = *(const u32x4*)g;
#endif
}
__device__ __forceinline__ void wait_async_le4() {
#ifdef GAT_ASYNC_LDS
  asm volatile("s_wait_asynccnt 0x4" ::: "memory");
#endif
}
__device__ __forceinline__ void wait_async_0() {
#ifdef GAT_ASYNC_LDS
  asm volatile("s_wait_asynccnt 0x0" ::: "memory");
#endif
}

// ---------- generic fill ----------
__global__ void gat_fill_u32(unsigned* p, long long n, unsigned v) {
  long long t = (long long)blockIdx.x * blockDim.x + threadIdx.x;
  if (t < n) p[t] = v;
}

// ---------- f32 -> f16 converts ----------
__global__ void gat_cvt_f16(const float* __restrict__ x, _Float16* __restrict__ xh, long long n) {
  long long t = (long long)blockIdx.x * blockDim.x + threadIdx.x;
  if (t < n) xh[t] = (_Float16)x[t];
}
// Wt[n][k] = (f16) W[k][n], 768x768
__global__ void gat_cvt_wT(const float* __restrict__ W, _Float16* __restrict__ Wt) {
  int t = blockIdx.x * blockDim.x + threadIdx.x;
  if (t >= NIN * HOUT) return;
  int n = t / NIN, k = t - n * NIN;
  Wt[t] = (_Float16)W[(size_t)k * HOUT + n];
}

// ---------- WMMA GEMM: Y[M,768] = Xh[M,768](f16) @ Wt^T + bias ----------
// Wt is [n][k] (pre-transposed f16 weights). 256 threads = 8 waves.
// Tile 128(M) x 128(N) x 32(K); wave w: wm=w&3 (32 rows), wn=w>>2 (64 cols).
// Double-buffered LDS, async global->LDS copies (ASYNCcnt pipelined).
__global__ __launch_bounds__(256)
void gat_gemm768(const _Float16* __restrict__ Xh, const _Float16* __restrict__ Wt,
                 const float* __restrict__ bias, float* __restrict__ Y, int M) {
  __shared__ _Float16 As[2][128][40];   // [buf][m][k]  (80B rows: 16B aligned)
  __shared__ _Float16 Bs[2][128][40];   // [buf][n][k]

  const int tid  = threadIdx.x;
  const int lane = tid & 31;
  const int wv   = tid >> 5;
  const int wm   = wv & 3;
  const int wn   = wv >> 2;
  const int mrow = lane & 15;
  const int hi   = lane >> 4;

  const int m0 = blockIdx.y * 128;
  const int n0 = blockIdx.x * 128;

  // per-thread copy slots: 4 rows of 8 halves (16B) per tile
  const int crow = tid >> 2;          // 0..63
  const int ccol = (tid & 3) * 8;     // 0,8,16,24

  v8f acc[2][4];
#pragma unroll
  for (int i = 0; i < 2; ++i)
#pragma unroll
    for (int j = 0; j < 4; ++j) acc[i][j] = (v8f)(0.0f);

  auto issue_tile = [&](int kt, int buf) {
    int k0 = kt * 32;
#pragma unroll
    for (int i = 0; i < 2; ++i) {
      int r  = crow + i * 64;
      int ga = min(m0 + r, M - 1);            // clamp: rows >= M never stored
      cp16_g2s(&As[buf][r][ccol], Xh + (size_t)ga * NIN + k0 + ccol);
      cp16_g2s(&Bs[buf][r][ccol], Wt + (size_t)(n0 + r) * NIN + k0 + ccol);
    }
  };

  issue_tile(0, 0);
  for (int kt = 0; kt < KTILES; ++kt) {
    const int cur = kt & 1;
    if (kt + 1 < KTILES) { issue_tile(kt + 1, cur ^ 1); wait_async_le4(); }
    else                 { wait_async_0(); }
    __syncthreads();

    v16h afr[2], bfr[4];
#pragma unroll
    for (int mi = 0; mi < 2; ++mi) {
      int m = wm * 32 + mi * 16 + mrow;
#pragma unroll
      for (int v = 0; v < 8; ++v) {
        int kb = (v < 4 ? 2 * v : 16 + 2 * (v - 4)) + 8 * hi;
        afr[mi][2 * v]     = As[cur][m][kb];
        afr[mi][2 * v + 1] = As[cur][m][kb + 1];
      }
    }
#pragma unroll
    for (int ni = 0; ni < 4; ++ni) {
      int n = wn * 64 + ni * 16 + mrow;
#pragma unroll
      for (int v = 0; v < 8; ++v) {
        int kb = (v < 4 ? 2 * v : 16 + 2 * (v - 4)) + 8 * hi;
        bfr[ni][2 * v]     = Bs[cur][n][kb];
        bfr[ni][2 * v + 1] = Bs[cur][n][kb + 1];
      }
    }
#pragma unroll
    for (int mi = 0; mi < 2; ++mi)
#pragma unroll
      for (int ni = 0; ni < 4; ++ni)
        acc[mi][ni] = __builtin_amdgcn_wmma_f32_16x16x32_f16(
            false, afr[mi], false, bfr[ni], (short)0, acc[mi][ni], false, false);
    __syncthreads();
  }

  // store C (+bias). VGPR r -> row r + 8*hi, lane&15 -> col
#pragma unroll
  for (int mi = 0; mi < 2; ++mi)
#pragma unroll
    for (int ni = 0; ni < 4; ++ni)
#pragma unroll
      for (int r = 0; r < 8; ++r) {
        int gr = m0 + wm * 32 + mi * 16 + r + 8 * hi;
        int gc = n0 + wn * 64 + ni * 16 + mrow;
        if (gr < M) Y[(size_t)gr * HOUT + gc] = acc[mi][ni][r] + bias[gc];
      }
}

// ---------- layer-1 edge kernels ----------
__global__ void gat_logits1(const int* ei, const float* xl, const float* xr,
                            const float* att1, float* lg, unsigned* smax,
                            long long E, long long Etot) {
  long long t = (long long)blockIdx.x * blockDim.x + threadIdx.x;
  if (t >= Etot * HEADS) return;
  long long e = t >> 3;
  int h = (int)(t & 7);
  int s, d; edge_sd(ei, e, E, s, d);
  const float* pl = xl + (size_t)s * HOUT + h * HID;
  const float* pr = xr + (size_t)d * HOUT + h * HID;
  const float* pa = att1 + h * HID;
  float acc = 0.0f;
#pragma unroll 4
  for (int c = 0; c < HID; ++c) {
    float v = pl[c] + pr[c];
    v = (v > 0.0f) ? v : NEG_SLOPE * v;
    acc += v * pa[c];
  }
  lg[e * HEADS + h] = acc;
  atomicMax(&smax[(size_t)d * HEADS + h], fenc(acc));
}

__global__ void gat_exp1(const int* ei, float* lg, const unsigned* smax,
                         float* ssum, long long E, long long Etot) {
  long long t = (long long)blockIdx.x * blockDim.x + threadIdx.x;
  if (t >= Etot * HEADS) return;
  long long e = t >> 3;
  int h = (int)(t & 7);
  int s, d; edge_sd(ei, e, E, s, d);
  float m  = fdec(smax[(size_t)d * HEADS + h]);
  float ex = expf(lg[e * HEADS + h] - m);
  lg[e * HEADS + h] = ex;
  atomicAdd(&ssum[(size_t)d * HEADS + h], ex);
}

// grid = Etot*3 blocks of 256: 3 blocks cover one edge's 768 features
__global__ void gat_agg1(const int* ei, const float* xl, const float* lg,
                         const float* ssum, float* hacc, long long E) {
  unsigned b = blockIdx.x;
  long long e = b / 3u;
  int idx = (int)(b % 3u) * 256 + threadIdx.x;
  int h = idx / HID;
  int s, d; edge_sd(ei, e, E, s, d);
  float alpha = lg[e * HEADS + h] / (ssum[(size_t)d * HEADS + h] + 1e-16f);
  atomicAdd(&hacc[(size_t)d * HOUT + idx], alpha * xl[(size_t)s * HOUT + idx]);
}

// grid = Nn*3 blocks of 256
__global__ void gat_bias_elu(float* h, const float* bias1) {
  unsigned b = blockIdx.x;
  unsigned node = b / 3u;
  int c = (int)(b % 3u) * 256 + threadIdx.x;
  size_t t = (size_t)node * HOUT + c;
  float v = h[t] + bias1[c];
  h[t] = (v > 0.0f) ? v : (expf(v) - 1.0f);
}

// ---------- layer-2 node transform: one wave per node ----------
__global__ void gat_xform2(const float* __restrict__ h,
                           const float* __restrict__ W2l, const float* __restrict__ b2l,
                           const float* __restrict__ W2r, const float* __restrict__ b2r,
                           float* x2l, float* x2r, long long Nn) {
  long long gt = (long long)blockIdx.x * blockDim.x + threadIdx.x;
  long long node = gt >> 5;
  int lane = threadIdx.x & 31;
  if (node >= Nn) return;
  const float* hp = h + (size_t)node * HOUT;
  float al[NCLS] = {0.f, 0.f, 0.f};
  float ar[NCLS] = {0.f, 0.f, 0.f};
#pragma unroll 4
  for (int it = 0; it < HOUT / 32; ++it) {
    int j = it * 32 + lane;
    float hv = hp[j];
#pragma unroll
    for (int k = 0; k < NCLS; ++k) {
      al[k] += hv * W2l[j * NCLS + k];
      ar[k] += hv * W2r[j * NCLS + k];
    }
  }
#pragma unroll
  for (int off = 16; off > 0; off >>= 1) {
#pragma unroll
    for (int k = 0; k < NCLS; ++k) {
      al[k] += __shfl_down(al[k], off, 32);
      ar[k] += __shfl_down(ar[k], off, 32);
    }
  }
  if (lane == 0) {
#pragma unroll
    for (int k = 0; k < NCLS; ++k) {
      x2l[node * NCLS + k] = al[k] + b2l[k];
      x2r[node * NCLS + k] = ar[k] + b2r[k];
    }
  }
}

// ---------- layer-2 edge kernels ----------
__global__ void gat_logits2(const int* ei, const float* x2l, const float* x2r,
                            const float* att2, float* lg, unsigned* smax,
                            long long E, long long Etot) {
  long long e = (long long)blockIdx.x * blockDim.x + threadIdx.x;
  if (e >= Etot) return;
  int s, d; edge_sd(ei, e, E, s, d);
  float acc = 0.0f;
#pragma unroll
  for (int k = 0; k < NCLS; ++k) {
    float v = x2l[(size_t)s * NCLS + k] + x2r[(size_t)d * NCLS + k];
    v = (v > 0.0f) ? v : NEG_SLOPE * v;
    acc += v * att2[k];
  }
  lg[e] = acc;
  atomicMax(&smax[d], fenc(acc));
}

__global__ void gat_exp2(const int* ei, float* lg, const unsigned* smax,
                         float* ssum, long long E, long long Etot) {
  long long e = (long long)blockIdx.x * blockDim.x + threadIdx.x;
  if (e >= Etot) return;
  int s, d; edge_sd(ei, e, E, s, d);
  float ex = expf(lg[e] - fdec(smax[d]));
  lg[e] = ex;
  atomicAdd(&ssum[d], ex);
}

__global__ void gat_agg2(const int* ei, const float* x2l, const float* lg,
                         const float* ssum, float* o2, long long E, long long Etot) {
  long long e = (long long)blockIdx.x * blockDim.x + threadIdx.x;
  if (e >= Etot) return;
  int s, d; edge_sd(ei, e, E, s, d);
  float alpha = lg[e] / (ssum[d] + 1e-16f);
#pragma unroll
  for (int k = 0; k < NCLS; ++k)
    atomicAdd(&o2[(size_t)d * NCLS + k], alpha * x2l[(size_t)s * NCLS + k]);
}

__global__ void gat_final(const float* o2, const float* bias2, float* out, long long n) {
  long long t = (long long)blockIdx.x * blockDim.x + threadIdx.x;
  if (t >= n) return;
  out[t] = o2[t] + bias2[t % NCLS];
}

// ---------- host ----------
static inline long long cdiv(long long a, long long b) { return (a + b - 1) / b; }

extern "C" void kernel_launch(void* const* d_in, const int* in_sizes, int n_in,
                              void* d_out, int out_size, void* d_ws, size_t ws_size,
                              hipStream_t stream) {
  const float* x     = (const float*)d_in[0];
  const int*   ei    = (const int*)d_in[1];
  const float* W1l   = (const float*)d_in[2];
  const float* b1l   = (const float*)d_in[3];
  const float* W1r   = (const float*)d_in[4];
  const float* b1r   = (const float*)d_in[5];
  const float* att1  = (const float*)d_in[6];
  const float* bias1 = (const float*)d_in[7];
  const float* W2l   = (const float*)d_in[8];
  const float* b2l   = (const float*)d_in[9];
  const float* W2r   = (const float*)d_in[10];
  const float* b2r   = (const float*)d_in[11];
  const float* att2  = (const float*)d_in[12];
  const float* bias2 = (const float*)d_in[13];
  float* out = (float*)d_out;

  const long long Nn   = in_sizes[0] / NIN;
  const long long E    = in_sizes[1] / 2;
  const long long Etot = E + Nn;

  // carve workspace
  char* p = (char*)d_ws;
  auto take = [&](size_t bytes) {
    void* r = (void*)p;
    p += (bytes + 255) & ~(size_t)255;
    return r;
  };
  float*     xl   = (float*)take((size_t)Nn * HOUT * 4);
  float*     xr   = (float*)take((size_t)Nn * HOUT * 4);
  float*     hbuf = (float*)take((size_t)Nn * HOUT * 4);
  _Float16*  xh   = (_Float16*)take((size_t)Nn * NIN * 2);
  _Float16*  w1lt = (_Float16*)take((size_t)NIN * HOUT * 2);
  _Float16*  w1rt = (_Float16*)take((size_t)NIN * HOUT * 2);
  float*     lg1  = (float*)take((size_t)Etot * HEADS * 4);
  unsigned*  sm1  = (unsigned*)take((size_t)Nn * HEADS * 4);
  float*     ss1  = (float*)take((size_t)Nn * HEADS * 4);
  float*     x2l  = (float*)take((size_t)Nn * NCLS * 4);
  float*     x2r  = (float*)take((size_t)Nn * NCLS * 4);
  float*     lg2  = (float*)take((size_t)Etot * 4);
  unsigned*  sm2  = (unsigned*)take((size_t)Nn * 4);
  float*     ss2  = (float*)take((size_t)Nn * 4);
  float*     o2   = (float*)take((size_t)Nn * NCLS * 4);
  (void)ws_size; (void)n_in; (void)out_size;

  const int B = 256;
  // init
  gat_fill_u32<<<cdiv(Nn * HEADS, B), B, 0, stream>>>(sm1, Nn * HEADS, ENC_NEG_INF);
  gat_fill_u32<<<cdiv(Nn * HEADS, B), B, 0, stream>>>((unsigned*)ss1, Nn * HEADS, 0u);
  gat_fill_u32<<<cdiv(Nn * HOUT, B), B, 0, stream>>>((unsigned*)hbuf, Nn * HOUT, 0u);
  gat_fill_u32<<<cdiv(Nn, B), B, 0, stream>>>(sm2, Nn, ENC_NEG_INF);
  gat_fill_u32<<<cdiv(Nn, B), B, 0, stream>>>((unsigned*)ss2, Nn, 0u);
  gat_fill_u32<<<cdiv(Nn * NCLS, B), B, 0, stream>>>((unsigned*)o2, Nn * NCLS, 0u);

  // precision converts (X once; weights transposed to [n][k])
  gat_cvt_f16<<<cdiv(Nn * NIN, B), B, 0, stream>>>(x, xh, Nn * NIN);
  gat_cvt_wT<<<cdiv((long long)NIN * HOUT, B), B, 0, stream>>>(W1l, w1lt);
  gat_cvt_wT<<<cdiv((long long)NIN * HOUT, B), B, 0, stream>>>(W1r, w1rt);

  // layer-1 dense transforms (WMMA, async double-buffered tiles)
  dim3 ggrid(HOUT / 128, (unsigned)cdiv(Nn, 128));
  gat_gemm768<<<ggrid, B, 0, stream>>>(xh, w1lt, b1l, xl, (int)Nn);
  gat_gemm768<<<ggrid, B, 0, stream>>>(xh, w1rt, b1r, xr, (int)Nn);

  // layer-1 edge phase
  gat_logits1<<<cdiv(Etot * HEADS, B), B, 0, stream>>>(ei, xl, xr, att1, lg1, sm1, E, Etot);
  gat_exp1<<<cdiv(Etot * HEADS, B), B, 0, stream>>>(ei, lg1, sm1, ss1, E, Etot);
  gat_agg1<<<(unsigned)(Etot * 3), B, 0, stream>>>(ei, xl, lg1, ss1, hbuf, E);
  gat_bias_elu<<<(unsigned)(Nn * 3), B, 0, stream>>>(hbuf, bias1);

  // layer-2
  gat_xform2<<<cdiv(Nn * 32, B), B, 0, stream>>>(hbuf, W2l, b2l, W2r, b2r, x2l, x2r, Nn);
  gat_logits2<<<cdiv(Etot, B), B, 0, stream>>>(ei, x2l, x2r, att2, lg2, sm2, E, Etot);
  gat_exp2<<<cdiv(Etot, B), B, 0, stream>>>(ei, lg2, sm2, ss2, E, Etot);
  gat_agg2<<<cdiv(Etot, B), B, 0, stream>>>(ei, x2l, lg2, ss2, o2, E, Etot);
  gat_final<<<cdiv(Nn * NCLS, B), B, 0, stream>>>(o2, bias2, out, Nn * NCLS);
}
